// LightGCL_31147102830645
// MI455X (gfx1250) — matrix operand order repaired and checked by reading, not compiled, not run on hardware
//
#include <hip/hip_runtime.h>
#include <hip/hip_bf16.h>

typedef __attribute__((ext_vector_type(16))) _Float16 v16h;
typedef __attribute__((ext_vector_type(8)))  _Float16 v8h;
typedef __attribute__((ext_vector_type(8)))  float    v8f;

#define DDIM 64
#define RT   4   // row tiles (of 16) per wave in the WMMA kernel

// exp(dot/TEMP) = exp2(dot * (1/TEMP) * log2(e)); bake into A-side f16 scale
#define ASCALE 7.2134752044448170f   // 5.0 * log2(e)

// ---------------------------------------------------------------- elementwise

__global__ void mask_edges_kernel(const float* __restrict__ vals,
                                  const float* __restrict__ unif,
                                  float* __restrict__ out, int n)
{
    int i = blockIdx.x * blockDim.x + threadIdx.x;
    if (i < n) out[i] = (unif[i] >= 0.2f) ? vals[i] * 1.25f : 0.0f;
}

// copy src into three destinations, zero a fourth buffer
__global__ void init3_zero1_kernel(const float* __restrict__ src,
                                   float* __restrict__ d0, float* __restrict__ d1,
                                   float* __restrict__ d2, float* __restrict__ z0, int n)
{
    int i = blockIdx.x * blockDim.x + threadIdx.x;
    if (i < n) { float v = src[i]; d0[i] = v; d1[i] = v; d2[i] = v; z0[i] = 0.0f; }
}

__global__ void zero_kernel(float* __restrict__ p, int n)
{
    int i = blockIdx.x * blockDim.x + threadIdx.x;
    if (i < n) p[i] = 0.0f;
}

__global__ void accum_zero_kernel(float* __restrict__ sum, const float* __restrict__ nxt,
                                  float* __restrict__ tz, int n)
{
    int i = blockIdx.x * blockDim.x + threadIdx.x;
    if (i < n) { sum[i] += nxt[i]; tz[i] = 0.0f; }
}

__global__ void scale_conv_kernel(float* __restrict__ buf, _Float16* __restrict__ outh,
                                  int n, float s)
{
    int i = blockIdx.x * blockDim.x + threadIdx.x;
    if (i < n) {
        float v = buf[i] * s;
        buf[i] = v;
        if (outh) outh[i] = (_Float16)v;
    }
}

// gather rows src[idx[b]] -> dst, converting to f16 with an extra scale factor
__global__ void gather_half_kernel(const float* __restrict__ src, const int* __restrict__ idx,
                                   _Float16* __restrict__ dst, int nB, float scale)
{
    int t = blockIdx.x * blockDim.x + threadIdx.x;
    int b = t >> 6;
    if (b >= nB) return;
    int d = t & 63;
    dst[t] = (_Float16)(src[(size_t)idx[b] * DDIM + d] * scale);
}

// ---------------------------------------------------------------- SpMM (COO)
// out[rows[e], :] += vals[e] * x[cols[e], :]  -- 16 threads/edge, float4 per thread
__global__ void spmm_atomic_kernel(const int* __restrict__ rows,
                                   const int* __restrict__ cols,
                                   const float* __restrict__ vals,
                                   const float* __restrict__ x,
                                   float* __restrict__ out, int nE)
{
    int t = blockIdx.x * blockDim.x + threadIdx.x;
    int e = t >> 4;
    if (e >= nE) return;
    float v = vals[e];
    if (v == 0.0f) return;                 // dropped edge: no traffic
    int d0 = (t & 15) * 4;
    const float4 xv = *(const float4*)(x + (size_t)cols[e] * DDIM + d0);
    float* o = out + (size_t)rows[e] * DDIM + d0;
    atomicAdd(o + 0, v * xv.x);
    atomicAdd(o + 1, v * xv.y);
    atomicAdd(o + 2, v * xv.z);
    atomicAdd(o + 3, v * xv.w);
}

// ---------------------------------------------------------------- WMMA GEMM + exp row-sum
// A: [M x 64] f16 row-major (batch rows, PRE-SCALED by ASCALE),
// W: [N x 64] f16 row-major.
// rowsum[m] += sum_j exp2(dot(A[m], W[j]))  ( == sum_j exp(dot/TEMP) )
//
// Each wave owns RT row tiles (M = 16*RT) of A in registers and strides over
// the N/16 column tiles: each loaded B tile feeds 2*RT WMMAs.  The next
// column tile is prefetched (global_prefetch_b8) so the loads hit WGP$/L2.
__global__ void __launch_bounds__(256, 1)
wmma_expsum_kernel(const _Float16* __restrict__ A,
                   const _Float16* __restrict__ W,
                   float* __restrict__ rowsum,
                   int N)
{
    const int lane            = threadIdx.x & 31;
    const int wave_in_block   = threadIdx.x >> 5;
    const int waves_per_block = blockDim.x >> 5;
    const int jslice          = blockIdx.x * waves_per_block + wave_in_block;
    const int num_jslices     = gridDim.x * waves_per_block;
    const int row_base        = blockIdx.y * (16 * RT);

    const int n = lane & 15;   // row within A tile / column within B tile
    const int h = lane >> 4;   // lane half

    // ---- A fragments (registers; reused across all column tiles) ----
    // A operand, this code's contiguous k order: lane half h holds K = h*16+[0..15] (elems 0..15),
    // matching B.  Two K-blocks for K=64.
    v16h aK0[RT], aK1[RT];
#pragma unroll
    for (int r = 0; r < RT; ++r) {
        const _Float16* arow = A + (size_t)(row_base + r * 16 + n) * DDIM;
        aK0[r] = *(const v16h*)(arow +      h * 16);
        aK1[r] = *(const v16h*)(arow + 32 + h * 16);
    }

    float acc[RT][8];
#pragma unroll
    for (int r = 0; r < RT; ++r)
#pragma unroll
        for (int v = 0; v < 8; ++v) acc[r][v] = 0.0f;

    const int ntiles = N >> 4;

    // B operand, same contiguous k order as A: lane half h holds K = h*16+[0..15] of
    // column n.  B[k][n] = W[jbase+n][k] => contiguous row-major reads.
    for (int jt = jslice; jt < ntiles; jt += num_jslices) {
        const _Float16* wrow = W + (size_t)(jt * 16 + n) * DDIM;
        v16h b0 = *(const v16h*)(wrow +      h * 16);
        v16h b1 = *(const v16h*)(wrow + 32 + h * 16);

        int jn = jt + num_jslices;
        if (jn < ntiles) {   // pull next tile's line toward WGP$/L2
            __builtin_prefetch(W + (size_t)(jn * 16 + n) * DDIM, 0, 3);
        }

#pragma unroll
        for (int r = 0; r < RT; ++r) {
            v8f c = {};
            c = __builtin_amdgcn_wmma_f32_16x16x32_f16(false, aK0[r], false, b0,
                                                       (short)0, c, false, false);
            c = __builtin_amdgcn_wmma_f32_16x16x32_f16(false, aK1[r], false, b1,
                                                       (short)0, c, false, false);
            // C/D: elem v of lane = D[v + 8h][n]; scale already folded into A
#pragma unroll
            for (int v = 0; v < 8; ++v) acc[r][v] += exp2f(c[v]);
        }
    }

    // reduce over the 16 lanes (N dimension) inside each lane-half, accumulate
#pragma unroll
    for (int r = 0; r < RT; ++r) {
#pragma unroll
        for (int v = 0; v < 8; ++v) {
            float s = acc[r][v];
            s += __shfl_xor(s, 1);
            s += __shfl_xor(s, 2);
            s += __shfl_xor(s, 4);
            s += __shfl_xor(s, 8);
            acc[r][v] = s;
        }
        if (n == 0) {  // lanes 0 (rows 0..7) and 16 (rows 8..15)
            float* rs = rowsum + row_base + r * 16 + h * 8;
#pragma unroll
            for (int v = 0; v < 8; ++v) atomicAdd(rs + v, acc[r][v]);
        }
    }
}

// ---------------------------------------------------------------- pos scores + BPR
__global__ void scores_kernel(const float* __restrict__ ue, const float* __restrict__ ie,
                              const float* __restrict__ sue, const float* __restrict__ sie,
                              const int* __restrict__ users, const int* __restrict__ pos,
                              const int* __restrict__ neg, float* __restrict__ acc, int nB)
{
    int t = blockIdx.x * blockDim.x + threadIdx.x;
    int b = t >> 5;
    if (b >= nB) return;
    int l = t & 31;
    int u = users[b], p = pos[b], ng = neg[b];
    const float* ueu  = ue  + (size_t)u  * DDIM;
    const float* iep  = ie  + (size_t)p  * DDIM;
    const float* ien  = ie  + (size_t)ng * DDIM;
    const float* suu  = sue + (size_t)u  * DDIM;
    const float* sin2 = sie + (size_t)ng * DDIM;
    float d1 = 0, d2 = 0, d3 = 0, d4 = 0;
#pragma unroll
    for (int k = l; k < DDIM; k += 32) {
        float uv = ueu[k], iv = ien[k];
        d1 += suu[k] * uv;   // sue[u] . ue[u]
        d2 += uv * iep[k];   // ue[u] . ie[pos]
        d3 += uv * iv;       // ue[u] . ie[neg]
        d4 += sin2[k] * iv;  // sie[neg] . ie[neg]
    }
#pragma unroll
    for (int off = 16; off; off >>= 1) {
        d1 += __shfl_xor(d1, off);
        d2 += __shfl_xor(d2, off);
        d3 += __shfl_xor(d3, off);
        d4 += __shfl_xor(d4, off);
    }
    if (l == 0) {
        const float inv_temp = 5.0f;
        float pu = fminf(fmaxf(d1 * inv_temp, -5.0f), 5.0f);
        float pi = fminf(fmaxf(d4 * inv_temp, -5.0f), 5.0f);
        float x  = d3 - d2;
        float sp = fmaxf(x, 0.0f) + log1pf(__expf(-fabsf(x)));  // softplus
        atomicAdd(&acc[0], pu);
        atomicAdd(&acc[1], pi);
        atomicAdd(&acc[2], sp);
    }
}

__global__ void finalize_kernel(const float* __restrict__ rowsum_u,
                                const float* __restrict__ rowsum_i,
                                const float* __restrict__ acc,
                                float* __restrict__ out, int nB)
{
    __shared__ float s[256];
    float l = 0.0f;
    for (int b = threadIdx.x; b < nB; b += 256)
        l += logf(rowsum_u[b] + 1e-8f) + logf(rowsum_i[b] + 1e-8f);
    s[threadIdx.x] = l;
    __syncthreads();
    for (int st = 128; st > 0; st >>= 1) {
        if (threadIdx.x < st) s[threadIdx.x] += s[threadIdx.x + st];
        __syncthreads();
    }
    if (threadIdx.x == 0) {
        float invB      = 1.0f / (float)nB;
        float neg_score = s[0] * invB;
        float pos_score = (acc[0] + acc[1]) * invB;
        float bpr       = acc[2] * invB;
        out[0] = bpr + 0.2f * (-pos_score + neg_score);
    }
}

// ---------------------------------------------------------------- launcher

extern "C" void kernel_launch(void* const* d_in, const int* in_sizes, int n_in,
                              void* d_out, int out_size, void* d_ws, size_t ws_size,
                              hipStream_t stream)
{
    const float* user_emb = (const float*)d_in[0];
    const float* item_emb = (const float*)d_in[1];
    const float* g_vals   = (const float*)d_in[2];
    const float* svd_vals = (const float*)d_in[3];
    const float* g_unif   = (const float*)d_in[4];
    const float* svd_unif = (const float*)d_in[5];
    const int*   g_rows   = (const int*)d_in[6];
    const int*   g_cols   = (const int*)d_in[7];
    const int*   svd_rows = (const int*)d_in[8];
    const int*   svd_cols = (const int*)d_in[9];
    const int*   users    = (const int*)d_in[10];
    const int*   positems = (const int*)d_in[11];
    const int*   negitems = (const int*)d_in[12];

    const int U = in_sizes[0] / DDIM;
    const int I = in_sizes[1] / DDIM;
    const int E = in_sizes[2];
    const int B = in_sizes[10];
    const int nU = U * DDIM, nI = I * DDIM;

    // workspace carve (256B aligned)
    char* w = (char*)d_ws;
    size_t off = 0;
    auto carve = [&](size_t bytes) -> char* {
        char* p = w + off;
        off = (off + bytes + 255) & ~(size_t)255;
        return p;
    };
    float*    gv       = (float*)carve((size_t)E * 4);
    float*    sv       = (float*)carve((size_t)E * 4);
    float*    uA       = (float*)carve((size_t)nU * 4);
    float*    uB       = (float*)carve((size_t)nU * 4);
    float*    iA       = (float*)carve((size_t)nI * 4);
    float*    iB       = (float*)carve((size_t)nI * 4);
    float*    ue       = (float*)carve((size_t)nU * 4);
    float*    ie       = (float*)carve((size_t)nI * 4);
    float*    sue      = (float*)carve((size_t)nU * 4);
    float*    sie      = (float*)carve((size_t)nI * 4);
    _Float16* ue_h     = (_Float16*)carve((size_t)nU * 2);
    _Float16* ie_h     = (_Float16*)carve((size_t)nI * 2);
    _Float16* sueb_h   = (_Float16*)carve((size_t)B * DDIM * 2);
    _Float16* sieb_h   = (_Float16*)carve((size_t)B * DDIM * 2);
    float*    rowsum_u = (float*)carve((size_t)B * 4);
    float*    rowsum_i = (float*)carve((size_t)B * 4);
    float*    acc      = (float*)carve(64);
    (void)ws_size; (void)n_in; (void)out_size;

    const int TPB = 256;
    auto blk = [&](long n) { return (int)((n + TPB - 1) / TPB); };

    // 1) edge dropout masks
    mask_edges_kernel<<<blk(E), TPB, 0, stream>>>(g_vals, g_unif, gv, E);
    mask_edges_kernel<<<blk(E), TPB, 0, stream>>>(svd_vals, svd_unif, sv, E);

    // 2) init layer-0 state + accumulators, zero "next" buffers
    init3_zero1_kernel<<<blk(nU), TPB, 0, stream>>>(user_emb, uA, ue, sue, uB, nU);
    init3_zero1_kernel<<<blk(nI), TPB, 0, stream>>>(item_emb, iA, ie, sie, iB, nI);
    zero_kernel<<<blk(B), TPB, 0, stream>>>(rowsum_u, B);
    zero_kernel<<<blk(B), TPB, 0, stream>>>(rowsum_i, B);
    zero_kernel<<<1, 64, 0, stream>>>(acc, 16);

    // 3) two propagation layers (svd path accumulates straight into sue/sie)
    float* u_prev = uA; float* u_next = uB;
    float* i_prev = iA; float* i_next = iB;
    const int spmm_blocks = blk((long)E * 16);
    for (int layer = 0; layer < 2; ++layer) {
        spmm_atomic_kernel<<<spmm_blocks, TPB, 0, stream>>>(g_rows, g_cols, gv, i_prev, u_next, E);
        spmm_atomic_kernel<<<spmm_blocks, TPB, 0, stream>>>(g_cols, g_rows, gv, u_prev, i_next, E);
        spmm_atomic_kernel<<<spmm_blocks, TPB, 0, stream>>>(svd_rows, svd_cols, sv, i_prev, sue, E);
        spmm_atomic_kernel<<<spmm_blocks, TPB, 0, stream>>>(svd_cols, svd_rows, sv, u_prev, sie, E);
        accum_zero_kernel<<<blk(nU), TPB, 0, stream>>>(ue, u_next, u_prev, nU);
        accum_zero_kernel<<<blk(nI), TPB, 0, stream>>>(ie, i_next, i_prev, nI);
        float* t;
        t = u_prev; u_prev = u_next; u_next = t;
        t = i_prev; i_prev = i_next; i_next = t;
    }

    // 4) mean over layers (1/3), emit f16 copies of ue/ie for the WMMA GEMM
    const float s3 = 1.0f / 3.0f;
    scale_conv_kernel<<<blk(nU), TPB, 0, stream>>>(ue, ue_h, nU, s3);
    scale_conv_kernel<<<blk(nI), TPB, 0, stream>>>(ie, ie_h, nI, s3);
    scale_conv_kernel<<<blk(nU), TPB, 0, stream>>>(sue, (_Float16*)nullptr, nU, s3);
    scale_conv_kernel<<<blk(nI), TPB, 0, stream>>>(sie, (_Float16*)nullptr, nI, s3);

    // 5) gather batch rows as f16, pre-scaled by (1/TEMP)*log2(e) for exp2
    gather_half_kernel<<<blk((long)B * DDIM), TPB, 0, stream>>>(sue, users, sueb_h, B, ASCALE);
    gather_half_kernel<<<blk((long)B * DDIM), TPB, 0, stream>>>(sie, negitems, sieb_h, B, ASCALE);

    // 6) WMMA exp-sum GEMMs: [B x 64] @ [N x 64]^T, exp2, row-sum
    //    y: row groups of 64 batch rows; x*8 waves: column slices
    dim3 grid(16, B / (16 * RT));
    wmma_expsum_kernel<<<grid, 256, 0, stream>>>(sueb_h, ue_h, rowsum_u, U);
    wmma_expsum_kernel<<<grid, 256, 0, stream>>>(sieb_h, ie_h, rowsum_i, I);

    // 7) pos scores + BPR, then final scalar
    scores_kernel<<<blk((long)B * 32), TPB, 0, stream>>>(ue, ie, sue, sie, users,
                                                         positems, negitems, acc, B);
    finalize_kernel<<<1, 256, 0, stream>>>(rowsum_u, rowsum_i, acc, (float*)d_out, B);
}
